// GraphNN_62689342653103
// MI455X (gfx1250) — matrix operand order repaired and checked by reading, not compiled, not run on hardware
//
#include <hip/hip_runtime.h>

#define NN   50000
#define EE   800000
#define FIN  92
#define DD   64
#define EFD  50
#define LL   3
#define ZZ   178      // 2*D + EF
#define KPAD 192      // ZZ padded to multiple of 32
#define GG   128
#define KC_N 6        // KPAD/32
#define NT_N 4        // DD/16
#define EPS  1e-5f

typedef __attribute__((ext_vector_type(16))) __bf16         v16bf;
typedef __attribute__((ext_vector_type(2)))  __bf16         v2bf;
typedef __attribute__((ext_vector_type(8)))  float          v8f;
typedef __attribute__((ext_vector_type(16))) unsigned short v16u;
typedef __attribute__((ext_vector_type(8)))  unsigned short v8u;

__device__ __forceinline__ unsigned short f2bf(float f) {
    unsigned int u = __float_as_uint(f);
    unsigned int r = u + 0x7FFFu + ((u >> 16) & 1u);   // round-to-nearest-even
    return (unsigned short)(r >> 16);
}

// Pack two fp32 -> one dword of two bf16. Prefer the single-instruction
// packed convert when the toolchain declares it for gfx1250.
__device__ __forceinline__ unsigned pack2bf(float a, float b) {
#if __has_builtin(__builtin_amdgcn_cvt_pk_bf16_f32)
    v2bf p = __builtin_amdgcn_cvt_pk_bf16_f32(a, b);
    return __builtin_bit_cast(unsigned, p);
#else
    return (unsigned)f2bf(a) | ((unsigned)f2bf(b) << 16);
#endif
}

// sigmoid(x)*softplus(x2) built purely on v_exp_f32 / v_log_f32 / v_rcp_f32,
// no libm inlining. softplus: max(x,0) + log(1 + e^-|x|)  (stable form).
__device__ __forceinline__ float gate_act(float xf, float xs) {
    float sig = __builtin_amdgcn_rcpf(1.0f + __expf(-xf));
    float sp  = fmaxf(xs, 0.0f) + __logf(1.0f + __expf(-fabsf(xs)));
    return sig * sp;
}

// ---------------------------------------------------------------------------
// Pre-swizzle Wf/Ws (fp32 [L,Z,D]) into bf16 WMMA-B-fragment layout:
// wswz[ ((((l*2+mat)*KC_N+kc)*NT_N+nt)*32 + lane)*16 + i ]
// B (32x16, KxN): n = lane&15, k_local = half*8 + i (i<8), 16 + half*8 + (i-8) (i>=8)
// ---------------------------------------------------------------------------
__global__ void k_prep(const float* __restrict__ Wf, const float* __restrict__ Ws,
                       unsigned short* __restrict__ wswz) {
    int idx = blockIdx.x * blockDim.x + threadIdx.x;
    const int total = LL * 2 * KC_N * NT_N * 32 * 16;
    if (idx >= total) return;
    int i    = idx & 15;  int t = idx >> 4;
    int lane = t & 31;    t >>= 5;
    int nt   = t & 3;     t >>= 2;
    int kc   = t % KC_N;  t /= KC_N;
    int mat  = t & 1;
    int l    = t >> 1;
    int k = kc * 32 + (lane >> 4) * 8 + (i & 7) + (i >> 3) * 16;
    int n = nt * 16 + (lane & 15);
    float v = 0.0f;
    if (k < ZZ) {
        const float* W = mat ? Ws : Wf;
        v = W[((size_t)l * ZZ + k) * DD + n];
    }
    wswz[idx] = f2bf(v);
}

// ---------------------------------------------------------------------------
// h = relu(x @ pre_W + pre_b)   x:[N,92]  pre_W:[92,64]
// ---------------------------------------------------------------------------
__global__ void k_pre(const float* __restrict__ x, const float* __restrict__ W,
                      const float* __restrict__ b, float* __restrict__ h) {
    __shared__ float xs[4][FIN];
    int t  = threadIdx.x;
    int ln = t >> 6;          // local node 0..3
    int d  = t & 63;
    int n  = blockIdx.x * 4 + ln;
    if (n < NN) {
        if (d < FIN)      xs[ln][d]      = x[(size_t)n * FIN + d];
        if (d + 64 < FIN) xs[ln][d + 64] = x[(size_t)n * FIN + d + 64];
    }
    __syncthreads();
    if (n < NN) {
        float acc = b[d];
        #pragma unroll 4
        for (int k = 0; k < FIN; ++k) acc = fmaf(xs[ln][k], W[k * DD + d], acc);
        h[(size_t)n * DD + d] = fmaxf(acc, 0.0f);
    }
}

__global__ void k_deg(const int* __restrict__ dst, float* __restrict__ deg) {
    int e = blockIdx.x * blockDim.x + threadIdx.x;
    if (e < EE) atomicAdd(&deg[dst[e]], 1.0f);
}

__global__ void k_invdeg(const float* __restrict__ deg, float* __restrict__ inv) {
    int n = blockIdx.x * blockDim.x + threadIdx.x;
    if (n < NN) inv[n] = 1.0f / fmaxf(deg[n], 1.0f);
}

// ---------------------------------------------------------------------------
// Edge message kernel: per 16-edge tile per wave, build z=[h_i||h_j||e] in LDS
// (bf16, K padded to 192), then 2 gates x 4 N-tiles x 6 K-chunks WMMAs,
// sigmoid*softplus, atomic scatter-add into agg[dst].
// ---------------------------------------------------------------------------
__global__ void __launch_bounds__(256)
k_edge(const float* __restrict__ h, const int* __restrict__ src,
       const int* __restrict__ dst, const float* __restrict__ ea,
       const unsigned short* __restrict__ wswz_l,
       const float* __restrict__ bfv, const float* __restrict__ bsv,
       float* __restrict__ agg) {
    __shared__ __align__(16) unsigned short As[8 * 16 * KPAD];

    int t    = threadIdx.x;
    int wave = t >> 5;
    int lane = t & 31;
    int e0   = (blockIdx.x * 8 + wave) * 16;

    // ---- stage A tile: lane pair (r, half) covers 96 cols of row r ----
    int r       = lane & 15;
    int halfsel = lane >> 4;
    int e       = e0 + r;
    bool valid  = (e < EE);
    int sidx = 0, didx = 0;
    if (valid) { sidx = src[e]; didx = dst[e]; }
    const float* hD  = h + (size_t)didx * DD;
    const float* hS  = h + (size_t)sidx * DD;
    const float* eap = ea + (size_t)e * EFD;   // 8-byte aligned (e*200 bytes)
    unsigned short* rowp = &As[(wave * 16 + r) * KPAD];
    int cbase = halfsel * 96;
    #pragma unroll
    for (int c4 = 0; c4 < 24; ++c4) {
        int c = cbase + c4 * 4;
        float f0, f1, f2, f3;
        if (!valid) {
            f0 = f1 = f2 = f3 = 0.0f;
        } else if (c < 64) {
            float4 v = *(const float4*)(hD + c);
            f0 = v.x; f1 = v.y; f2 = v.z; f3 = v.w;
        } else if (c < 128) {
            float4 v = *(const float4*)(hS + (c - 64));
            f0 = v.x; f1 = v.y; f2 = v.z; f3 = v.w;
        } else if (c < 176) {                   // ea elems a0..a0+3 all valid
            int a0 = c - 128;
            float2 u = *(const float2*)(eap + a0);
            float2 w = *(const float2*)(eap + a0 + 2);
            f0 = u.x; f1 = u.y; f2 = w.x; f3 = w.y;
        } else if (c == 176) {                  // elems 48,49 then zero pad
            float2 u = *(const float2*)(eap + 48);
            f0 = u.x; f1 = u.y; f2 = 0.0f; f3 = 0.0f;
        } else {                                // c >= 178: zero pad
            f0 = f1 = f2 = f3 = 0.0f;
        }
        uint2 p;
        p.x = pack2bf(f0, f1);
        p.y = pack2bf(f2, f3);
        *(uint2*)(rowp + c) = p;
    }
    __syncthreads();

    // ---- load A fragments: 2 x ds_load_b128 per 32-K chunk ----
    // ISA 16-bit A layout: lanes0-15 K in {half*8..+7, 16+half*8..+7} per chunk
    v16u a[KC_N];
    const unsigned short* arow = &As[(wave * 16 + (lane & 15)) * KPAD];
    int half8 = (lane >> 4) * 8;
    #pragma unroll
    for (int kc = 0; kc < KC_N; ++kc) {
        v8u lo = *(const v8u*)(arow + kc * 32 + half8);
        v8u hi = *(const v8u*)(arow + kc * 32 + 16 + half8);
        v16u av;
        #pragma unroll
        for (int i = 0; i < 8; ++i) { av[i] = lo[i]; av[i + 8] = hi[i]; }
        a[kc] = av;
    }

    // dst ids for the 8 rows this lane will scatter (C/D layout: m = r + 8*half)
    int mbase = (lane >> 4) * 8;
    int dstid[8];
    #pragma unroll
    for (int q = 0; q < 8; ++q) {
        int eq = e0 + mbase + q;
        dstid[q] = (eq < EE) ? dst[eq] : 0;
    }
    int ncol = lane & 15;

    #pragma unroll
    for (int nt = 0; nt < NT_N; ++nt) {
        float bfb = bfv[nt * 16 + ncol];
        float bsb = bsv[nt * 16 + ncol];
        v8f accf, accs;
        #pragma unroll
        for (int i = 0; i < 8; ++i) { accf[i] = bfb; accs[i] = bsb; }

        #pragma unroll
        for (int kc = 0; kc < KC_N; ++kc) {
            const unsigned short* bpf =
                wswz_l + ((size_t)((0 * KC_N + kc) * NT_N + nt) * 512) + lane * 16;
            const unsigned short* bps =
                wswz_l + ((size_t)((1 * KC_N + kc) * NT_N + nt) * 512) + lane * 16;
            v8u fl = *(const v8u*)bpf, fh = *(const v8u*)(bpf + 8);
            v8u sl = *(const v8u*)bps, sh = *(const v8u*)(bps + 8);
            v16u bfr, bsr;
            #pragma unroll
            for (int i = 0; i < 8; ++i) {
                bfr[i] = fl[i]; bfr[i + 8] = fh[i];
                bsr[i] = sl[i]; bsr[i + 8] = sh[i];
            }
            accf = __builtin_amdgcn_wmma_f32_16x16x32_bf16(
                false, __builtin_bit_cast(v16bf, a[kc]),
                false, __builtin_bit_cast(v16bf, bfr),
                (short)0, accf, false, false);
            accs = __builtin_amdgcn_wmma_f32_16x16x32_bf16(
                false, __builtin_bit_cast(v16bf, a[kc]),
                false, __builtin_bit_cast(v16bf, bsr),
                (short)0, accs, false, false);
        }

        #pragma unroll
        for (int q = 0; q < 8; ++q) {
            float mv = gate_act(accf[q], accs[q]);
            int eq = e0 + mbase + q;
            if (eq < EE)
                atomicAdd(&agg[(size_t)dstid[q] * DD + nt * 16 + ncol], mv);
        }
    }
}

// ---------------------------------------------------------------------------
// h = BN( h + agg * inv_deg )
// ---------------------------------------------------------------------------
__global__ void k_node(float* __restrict__ h, const float* __restrict__ agg,
                       const float* __restrict__ invd,
                       const float* __restrict__ gam, const float* __restrict__ bet,
                       const float* __restrict__ mu,  const float* __restrict__ var) {
    int idx = blockIdx.x * blockDim.x + threadIdx.x;
    if (idx >= NN * DD) return;
    int n = idx >> 6, d = idx & 63;
    float v = h[idx] + agg[idx] * invd[n];
    h[idx] = gam[d] * (v - mu[d]) * rsqrtf(var[d] + EPS) + bet[d];
}

__global__ void k_pool(const float* __restrict__ h, const int* __restrict__ batch,
                       float* __restrict__ pool, float* __restrict__ cnt) {
    int idx = blockIdx.x * blockDim.x + threadIdx.x;
    if (idx >= NN * DD) return;
    int n = idx >> 6, d = idx & 63;
    int b = batch[n];
    atomicAdd(&pool[(size_t)b * DD + d], h[idx]);
    if (d == 0) atomicAdd(&cnt[b], 1.0f);
}

__global__ void k_head(const float* __restrict__ pool, const float* __restrict__ cnt,
                       const float* __restrict__ pW, const float* __restrict__ pb,
                       const float* __restrict__ oW, const float* __restrict__ ob,
                       float* __restrict__ out) {
    int g = threadIdx.x;
    if (g >= GG) return;
    float row[DD];
    float ic = 1.0f / fmaxf(cnt[g], 1.0f);
    #pragma unroll
    for (int d = 0; d < DD; ++d) row[d] = pool[(size_t)g * DD + d] * ic;
    float o = ob[0];
    for (int j = 0; j < DD; ++j) {
        float acc = pb[j];
        #pragma unroll
        for (int d = 0; d < DD; ++d) acc = fmaf(row[d], pW[d * DD + j], acc);
        o = fmaf(fmaxf(acc, 0.0f), oW[j], o);
    }
    out[g] = o;
}

// ---------------------------------------------------------------------------
extern "C" void kernel_launch(void* const* d_in, const int* in_sizes, int n_in,
                              void* d_out, int out_size, void* d_ws, size_t ws_size,
                              hipStream_t stream) {
    (void)in_sizes; (void)n_in; (void)out_size; (void)ws_size;
    const float* x    = (const float*)d_in[0];
    const int*   ei   = (const int*)  d_in[1];
    const float* ea   = (const float*)d_in[2];
    const int*   bat  = (const int*)  d_in[3];
    const float* preW = (const float*)d_in[4];
    const float* preB = (const float*)d_in[5];
    const float* Wf   = (const float*)d_in[6];
    const float* bf   = (const float*)d_in[7];
    const float* Ws   = (const float*)d_in[8];
    const float* bs   = (const float*)d_in[9];
    const float* gam  = (const float*)d_in[10];
    const float* bet  = (const float*)d_in[11];
    const float* mu   = (const float*)d_in[12];
    const float* var  = (const float*)d_in[13];
    const float* pW   = (const float*)d_in[14];
    const float* pb   = (const float*)d_in[15];
    const float* oW   = (const float*)d_in[16];
    const float* ob   = (const float*)d_in[17];
    const int* src  = ei;          // edge_index[0]
    const int* dstp = ei + EE;     // edge_index[1]

    char* ws = (char*)d_ws;
    float* h    = (float*)ws;  ws += (size_t)NN * DD * 4;
    float* agg  = (float*)ws;  ws += (size_t)NN * DD * 4;
    float* deg  = (float*)ws;  ws += ((size_t)NN * 4 + 255) / 256 * 256;
    float* invd = (float*)ws;  ws += ((size_t)NN * 4 + 255) / 256 * 256;
    unsigned short* wswz = (unsigned short*)ws;
    const size_t wswz_elems = (size_t)LL * 2 * KC_N * NT_N * 32 * 16;  // 73728
    ws += (wswz_elems * 2 + 255) / 256 * 256;
    float* pool = (float*)ws;  ws += (size_t)GG * DD * 4;
    float* cnt  = (float*)ws;  ws += (size_t)GG * 4;

    hipMemsetAsync(deg, 0, (size_t)NN * 4, stream);
    k_prep<<<(int)((wswz_elems + 255) / 256), 256, 0, stream>>>(Wf, Ws, wswz);
    k_pre<<<(NN + 3) / 4, 256, 0, stream>>>(x, preW, preB, h);
    k_deg<<<(EE + 255) / 256, 256, 0, stream>>>(dstp, deg);
    k_invdeg<<<(NN + 255) / 256, 256, 0, stream>>>(deg, invd);

    for (int l = 0; l < LL; ++l) {
        hipMemsetAsync(agg, 0, (size_t)NN * DD * 4, stream);
        k_edge<<<EE / 128, 256, 0, stream>>>(
            h, src, dstp, ea,
            wswz + (size_t)l * 2 * KC_N * NT_N * 512,
            bf + l * DD, bs + l * DD, agg);
        k_node<<<(NN * DD + 255) / 256, 256, 0, stream>>>(
            h, agg, invd, gam + l * DD, bet + l * DD, mu + l * DD, var + l * DD);
    }

    hipMemsetAsync(pool, 0, (size_t)GG * DD * 4, stream);
    hipMemsetAsync(cnt, 0, (size_t)GG * 4, stream);
    k_pool<<<(NN * DD + 255) / 256, 256, 0, stream>>>(h, bat, pool, cnt);
    k_head<<<1, 128, 0, stream>>>(pool, cnt, pW, pb, oW, ob, (float*)d_out);
}
